// Attention_DPSABlock_23098334118149
// MI455X (gfx1250) — matrix operand order repaired and checked by reading, not compiled, not run on hardware
//
#include <hip/hip_runtime.h>
#include <hip/hip_bf16.h>
#include <math.h>

typedef __bf16 bf16;
typedef __attribute__((ext_vector_type(16))) __bf16 v16bf;
typedef __attribute__((ext_vector_type(8)))  float  v8f;

#define DEV static __device__ __forceinline__

// ---------------------------------------------------------------- WMMA helpers
DEV v8f wmma_bf16(v16bf a, v16bf b, v8f c) {
  // D = A(16x32 bf16) * B(32x16 bf16) + C(16x16 f32)
  return __builtin_amdgcn_wmma_f32_16x16x32_bf16(false, a, false, b, (short)0, c,
                                                 false, false);
}

// Load a 16-element bf16 fragment as two 16-byte LDS reads.
DEV v16bf ldfrag(const bf16* p0, const bf16* p1) {
  union { uint4 u[2]; v16bf v; } u;
  u.u[0] = *(const uint4*)p0;
  u.u[1] = *(const uint4*)p1;
  return u.v;
}

// Async DMA: global (saddr + 32-bit voff) -> LDS byte offset ldsOff, 16 bytes/lane.
DEV void async_g2l_b128(unsigned ldsOff, unsigned gOff, const void* sbase) {
  asm volatile("global_load_async_to_lds_b128 %0, %1, %2"
               :: "v"(ldsOff), "v"(gOff), "s"(sbase) : "memory");
}
DEV void wait_async0() { asm volatile("s_wait_asynccnt 0" ::: "memory"); }

DEV float geluf(float v) { return 0.5f * v * (1.f + erff(v * 0.70710678118654752f)); }
DEV float sigmf(float v) { return 1.f / (1.f + __expf(-v)); }

// ---------------------------------------------------------------- f32 -> bf16 weight convert
__global__ __launch_bounds__(256)
void k_cvt(const float* __restrict__ s, bf16* __restrict__ d, int n) {
  int i = blockIdx.x * 256 + threadIdx.x;
  if (i < n) d[i] = (bf16)s[i];
}

// ---------------------------------------------------------------- conv1x1 GEMM
// out[b][outChOff+m][px] = resid + rscale[m]*( W[m,:]·act[b][:, px]*actScale + bias[m] )
// W: bf16 [M][K] (pre-converted); act: bf16 [B][K][HW].
// Block tile: all M rows x 128 pixels; 8 waves. Weights staged via async DMA.
__global__ __launch_bounds__(256)
void k_gemm(const bf16* __restrict__ W, const float* __restrict__ bias,
            const bf16* __restrict__ act, const float* __restrict__ actScale,
            float* __restrict__ out32, bf16* __restrict__ outbf,
            const float* __restrict__ resid, const float* __restrict__ rscale,
            int M, int K, int HW, int tilesPerB, int outCH, int outChOff) {
  extern __shared__ char smem_raw[];
  const int SW = K + 8;
  bf16* ldsW = (bf16*)smem_raw;             // [M][SW]   (LDS offset 0)
  bf16* ldsA = ldsW + (size_t)M * SW;       // [128][SW] pixel-major
  const unsigned ldsABase = (unsigned)((size_t)M * SW * 2);
  const int tid = threadIdx.x;
  const int b = blockIdx.x / tilesPerB;
  const int pix0 = (blockIdx.x % tilesPerB) * 128;

  // weights: async DMA, 16B (8 bf16) per lane per issue; M*K/8 is a multiple of 256
  const int kc8 = K >> 3;
  for (int i = tid; i < (M * K) >> 3; i += 256) {
    int m = i / kc8, kc = (i - m * kc8) << 3;
    async_g2l_b128((unsigned)((m * SW + kc) * 2), (unsigned)(i << 4), (const void*)W);
  }
  // activations: transpose to pixel-major (+ optional per-channel SCA scale)
  const bf16* ab = act + (size_t)b * K * HW + pix0;
  for (int i = tid; i < K * 128; i += 256) {
    int k = i >> 7, px = i & 127;
    float v = (float)ab[(size_t)k * HW + px];
    if (actScale) v *= actScale[b * K + k];
    ldsA[px * SW + k] = (bf16)v;
  }
  (void)ldsABase;
  wait_async0();
  __syncthreads();

  const int wave = tid >> 5, lane = tid & 31;
  const int col = lane & 15, hi = lane >> 4;
  const int total = (M >> 4) * 8;  // (M/16) M-tiles x 8 N-tiles
  for (int t = wave; t < total; t += 8) {
    const int mt = t >> 3, nt = t & 7;
    v8f acc = {0.f, 0.f, 0.f, 0.f, 0.f, 0.f, 0.f, 0.f};
    const bf16* wr = &ldsW[(mt * 16 + col) * SW];
    const bf16* ar = &ldsA[(nt * 16 + col) * SW];
    for (int k0 = 0; k0 < K; k0 += 32) {
      v16bf af = ldfrag(wr + k0 + hi * 8, wr + k0 + 16 + hi * 8);
      v16bf bfr = ldfrag(ar + k0 + hi * 8, ar + k0 + 16 + hi * 8);
      acc = wmma_bf16(af, bfr, acc);
    }
    const int px = pix0 + nt * 16 + col;
#pragma unroll
    for (int v = 0; v < 8; ++v) {
      int m = mt * 16 + v + hi * 8;
      float val = acc[v] + (bias ? bias[m] : 0.f);
      size_t oi = ((size_t)b * outCH + outChOff + m) * HW + px;
      if (resid) val = resid[oi] + (rscale ? rscale[m] : 1.f) * val;
      if (out32) out32[oi] = val;
      if (outbf) outbf[oi] = (bf16)val;
    }
  }
}

// ---------------------------------------------------------------- channel LayerNorm
// Per-pixel LN over CH channels; optional concat (in0 = first half, in1 = rest).
template <typename TIN>
__global__ __launch_bounds__(256)
void k_lnch(const TIN* __restrict__ in0, const TIN* __restrict__ in1,
            const float* __restrict__ g, const float* __restrict__ bt,
            float eps, int CH, bf16* __restrict__ out, int HW) {
  const int gp = blockIdx.x * 256 + threadIdx.x;
  const int b = gp / HW, px = gp - b * HW;
  const int CH0 = in1 ? (CH >> 1) : CH;
  float s = 0.f, s2 = 0.f;
  for (int c = 0; c < CH; ++c) {
    float v = (c < CH0) ? (float)in0[((size_t)b * CH0 + c) * HW + px]
                        : (float)in1[((size_t)b * CH0 + (c - CH0)) * HW + px];
    s += v; s2 += v * v;
  }
  float mean = s / CH;
  float var = s2 / CH - mean * mean;
  float inv = rsqrtf(var + eps);
  for (int c = 0; c < CH; ++c) {
    float v = (c < CH0) ? (float)in0[((size_t)b * CH0 + c) * HW + px]
                        : (float)in1[((size_t)b * CH0 + (c - CH0)) * HW + px];
    out[((size_t)b * CH + c) * HW + px] = (bf16)((v - mean) * inv * g[c] + bt[c]);
  }
}

// ---------------------------------------------------------------- depthwise + gate
// t: bf16 [B][256][HW]; ch 0..127 local pre-gate (3x3), 128..255 global (5x5).
__global__ __launch_bounds__(256)
void k_dwgate(const bf16* __restrict__ t,
              const float* __restrict__ w3, const float* __restrict__ b3,
              const float* __restrict__ w5, const float* __restrict__ b5,
              bf16* __restrict__ gl, bf16* __restrict__ gg, int H, int W) {
  const int HW = H * W;
  const int c = blockIdx.y, b = blockIdx.z;
  const int p = blockIdx.x * 256 + threadIdx.x;
  const int y = p / W, x = p - y * W;
  const size_t tb = (size_t)b * 256 * HW;

  float a0 = 0.f, a1 = 0.f;
  const float* wl0 = w3 + c * 9;
  const float* wl1 = w3 + (c + 64) * 9;
  for (int dy = -1; dy <= 1; ++dy) {
    int yy = y + dy; if ((unsigned)yy >= (unsigned)H) continue;
    for (int dx = -1; dx <= 1; ++dx) {
      int xx = x + dx; if ((unsigned)xx >= (unsigned)W) continue;
      int wi = (dy + 1) * 3 + (dx + 1);
      size_t pi = (size_t)yy * W + xx;
      a0 += wl0[wi] * (float)t[tb + (size_t)c * HW + pi];
      a1 += wl1[wi] * (float)t[tb + (size_t)(c + 64) * HW + pi];
    }
  }
  gl[((size_t)b * 64 + c) * HW + p] = (bf16)((a0 + b3[c]) * (a1 + b3[c + 64]));

  float g0 = 0.f, g1 = 0.f;
  const float* wg0 = w5 + c * 25;
  const float* wg1 = w5 + (c + 64) * 25;
  for (int dy = -2; dy <= 2; ++dy) {
    int yy = y + dy; if ((unsigned)yy >= (unsigned)H) continue;
    for (int dx = -2; dx <= 2; ++dx) {
      int xx = x + dx; if ((unsigned)xx >= (unsigned)W) continue;
      int wi = (dy + 2) * 5 + (dx + 2);
      size_t pi = (size_t)yy * W + xx;
      g0 += wg0[wi] * (float)t[tb + (size_t)(128 + c) * HW + pi];
      g1 += wg1[wi] * (float)t[tb + (size_t)(192 + c) * HW + pi];
    }
  }
  gg[((size_t)b * 64 + c) * HW + p] = (bf16)((g0 + b5[c]) * (g1 + b5[c + 64]));
}

// ---------------------------------------------------------------- row mean
template <typename TIN>
__global__ __launch_bounds__(256)
void k_rowmean(const TIN* __restrict__ in, float* __restrict__ out, int HW) {
  __shared__ float red[256];
  const size_t base = (size_t)blockIdx.x * HW;
  float s = 0.f;
  for (int i = threadIdx.x; i < HW; i += 256) s += (float)in[base + i];
  red[threadIdx.x] = s;
  __syncthreads();
  for (int st = 128; st > 0; st >>= 1) {
    if (threadIdx.x < st) red[threadIdx.x] += red[threadIdx.x + st];
    __syncthreads();
  }
  if (threadIdx.x == 0) out[blockIdx.x] = red[0] / (float)HW;
}

// ---------------------------------------------------------------- SCA (1x1 on means)
__global__ __launch_bounds__(128)
void k_sca(const float* wl, const float* bl, const float* wg, const float* bg,
           const float* ml, const float* mg, float* sl, float* sg) {
  const int b = blockIdx.x, t = threadIdx.x;
  const int c = t & 63;
  const float* w = (t < 64) ? (wl + c * 64) : (wg + c * 64);
  const float* m = (t < 64) ? (ml + b * 64) : (mg + b * 64);
  float s = (t < 64) ? bl[c] : bg[c];
  for (int k = 0; k < 64; ++k) s += w[k] * m[k];
  if (t < 64) sl[b * 64 + c] = s; else sg[b * 64 + c] = s;
}

// ---------------------------------------------------------------- SE MLP (channel interaction)
__global__ __launch_bounds__(64)
void k_mlp_ci(const float* __restrict__ mn, const float* w1, const float* b1,
              const float* bng, const float* bnb, const float* w2, const float* b2,
              float* __restrict__ out) {
  __shared__ float hid[32];
  const int b = blockIdx.x, t = threadIdx.x;
  const float ibn = rsqrtf(1.0f + 1e-5f);
  if (t < 32) {
    float h = b1[t];
    for (int c = 0; c < 64; ++c) h += w1[t * 64 + c] * mn[b * 64 + c];
    hid[t] = geluf(h * ibn * bng[t] + bnb[t]);
  }
  __syncthreads();
  float o = b2[t];
  for (int j = 0; j < 32; ++j) o += w2[t * 32 + j] * hid[j];
  out[b * 64 + t] = o;
}

// ---------------------------------------------------------------- spatial interaction, cf *= sigmoid(MLP(xs)) in place
__global__ __launch_bounds__(256)
void k_si(const float* __restrict__ xs, const float* w1, const float* b1,
          const float* bng, const float* bnb, const float* w2, const float* b2,
          float* __restrict__ cf, int HW) {
  const int gp = blockIdx.x * 256 + threadIdx.x;
  const int b = gp / HW, px = gp - b * HW;
  const float ibn = rsqrtf(1.0f + 1e-5f);
  float xv[64];
  for (int c = 0; c < 64; ++c) xv[c] = xs[((size_t)b * 64 + c) * HW + px];
  float s = b2[0];
  for (int j = 0; j < 32; ++j) {
    float h = b1[j];
    for (int c = 0; c < 64; ++c) h += w1[j * 64 + c] * xv[c];
    s += w2[j] * geluf(h * ibn * bng[j] + bnb[j]);
  }
  float sg = sigmf(s);
  for (int c = 0; c < 64; ++c) cf[((size_t)b * 64 + c) * HW + px] *= sg;
}

// ---------------------------------------------------------------- SimpleGate
__global__ __launch_bounds__(256)
void k_gate(const bf16* __restrict__ in, bf16* __restrict__ out, int HW) {
  const int gp = blockIdx.x * 256 + threadIdx.x;
  const int px = gp % HW;
  const int t = gp / HW;
  const int c = t & 63, b = t >> 6;
  size_t ia = ((size_t)b * 128 + c) * HW + px;
  out[((size_t)b * 64 + c) * HW + px] = (bf16)((float)in[ia] * (float)in[ia + (size_t)64 * HW]);
}

// ---------------------------------------------------------------- window attention
// block = 64 threads (2 waves = 2 heads); grid = B*nH*nW*2 (head pairs).
// V staged via async DMA; sigmoid(ci) folded into output, SCALE folded into scores.
__global__ __launch_bounds__(64)
void k_attn(const bf16* __restrict__ qkv, const float* __restrict__ ci,
            const float* __restrict__ rpb, float* __restrict__ xs, int H, int W) {
  const int HW = H * W;
  const int nWx = W >> 3, nHy = H >> 3;
  const int wid = blockIdx.x >> 1;
  const int hp = blockIdx.x & 1;
  const int b = wid / (nHy * nWx);
  const int rem = wid - b * (nHy * nWx);
  const int wh = rem / nWx, ww = rem - wh * nWx;
  const int py0 = wh << 3, px0 = ww << 3;
  const int tid = threadIdx.x;
  const int hl = tid >> 5;
  const int h = hp * 2 + hl;
  const int lane = tid & 31;
  const int col = lane & 15, hi = lane >> 4;

  // dynamic LDS (base offset 0): vs | qs | ks | ss
  extern __shared__ char smem[];
  bf16* vs = (bf16*)smem;                      // [2][16][64]  hd x tokens
  bf16* qs = (bf16*)(smem + 4096);             // [2][64][32]  tokens x (16 | 16 zero)
  bf16* ks = (bf16*)(smem + 4096 + 8192);      // [2][64][32]
  float* ss = (float*)(smem + 4096 + 16384);   // [2][64][68]

  const bf16* qkvb = qkv + (size_t)b * 192 * HW;
  // V: async DMA, one 16B window-row (8 tokens) per lane per issue
  for (int i = lane; i < 128; i += 32) {
    int d = i >> 3, r = i & 7;
    unsigned ldsOff = (unsigned)((((hl * 16 + d) * 64) + r * 8) * 2);
    unsigned gOff = (unsigned)((((size_t)(128 + h * 16 + d) * HW) +
                                (size_t)(py0 + r) * W + px0) * 2);
    async_g2l_b128(ldsOff, gOff, (const void*)qkvb);
  }
  // Q, K: plain copies (scales folded into epilogues)
  for (int i = lane; i < 64 * 16; i += 32) {
    int tok = i >> 4, d = i & 15;
    int px = (py0 + (tok >> 3)) * W + px0 + (tok & 7);
    int c = h * 16 + d;
    int qi = (hl * 64 + tok) * 32 + d;
    qs[qi] = qkvb[(size_t)c * HW + px];        qs[qi + 16] = (bf16)0.f;
    ks[qi] = qkvb[(size_t)(64 + c) * HW + px]; ks[qi + 16] = (bf16)0.f;
  }
  wait_async0();
  __syncthreads();

  // S = SCALE * Q K^T + relpos bias
  for (int mt = 0; mt < 4; ++mt)
    for (int nt = 0; nt < 4; ++nt) {
      const bf16* qr = &qs[(hl * 64 + mt * 16 + col) * 32];
      const bf16* kr = &ks[(hl * 64 + nt * 16 + col) * 32];
      v16bf a = ldfrag(qr + hi * 8, qr + 16 + hi * 8);
      v16bf bb = ldfrag(kr + hi * 8, kr + 16 + hi * 8);
      v8f acc = {0.f, 0.f, 0.f, 0.f, 0.f, 0.f, 0.f, 0.f};
      acc = wmma_bf16(a, bb, acc);
      int m = nt * 16 + col;
#pragma unroll
      for (int v = 0; v < 8; ++v) {
        int n = mt * 16 + v + hi * 8;
        int dh = (n >> 3) - (m >> 3) + 7;
        int dw = (n & 7) - (m & 7) + 7;
        ss[(hl * 64 + n) * 68 + m] = 0.25f * acc[v] + rpb[(dh * 15 + dw) * 4 + h];
      }
    }
  // softmax (2 rows per lane)
  for (int r = lane; r < 64; r += 32) {
    float* row = &ss[(hl * 64 + r) * 68];
    float mx = -1e30f;
    for (int m2 = 0; m2 < 64; ++m2) mx = fmaxf(mx, row[m2]);
    float sum = 0.f;
    for (int m2 = 0; m2 < 64; ++m2) { float e = __expf(row[m2] - mx); row[m2] = e; sum += e; }
    float inv = 1.f / sum;
    for (int m2 = 0; m2 < 64; ++m2) row[m2] *= inv;
  }
  // O = (P V) * sigmoid(ci)   (window-reverse [B][C][H][W] layout)
  const float vsc = sigmf(ci[b * 64 + h * 16 + col]);
  for (int mt = 0; mt < 4; ++mt) {
    v8f acc = {0.f, 0.f, 0.f, 0.f, 0.f, 0.f, 0.f, 0.f};
    for (int k0 = 0; k0 < 64; k0 += 32) {
      const float* sr = &ss[(hl * 64 + mt * 16 + col) * 68 + k0];
      v16bf a;
#pragma unroll
      for (int i2 = 0; i2 < 8; ++i2) {
        a[i2] = (bf16)sr[hi * 8 + i2];
        a[8 + i2] = (bf16)sr[16 + hi * 8 + i2];
      }
      const bf16* vr = &vs[(hl * 16 + col) * 64 + k0];
      v16bf bb = ldfrag(vr + hi * 8, vr + 16 + hi * 8);
      acc = wmma_bf16(a, bb, acc);
    }
#pragma unroll
    for (int v = 0; v < 8; ++v) {
      int tok = mt * 16 + v + hi * 8;
      int px = (py0 + (tok >> 3)) * W + px0 + (tok & 7);
      xs[((size_t)b * 64 + h * 16 + col) * HW + px] = acc[v] * vsc;
    }
  }
}

// ================================================================ host launch
extern "C" void kernel_launch(void* const* d_in, const int* in_sizes, int n_in,
                              void* d_out, int out_size, void* d_ws, size_t ws_size,
                              hipStream_t stream) {
  (void)in_sizes; (void)n_in; (void)out_size; (void)ws_size;
  const int B = 8, H = 128, W = 256, HW = H * W;
  const size_t P = (size_t)B * HW;
  const int NPB = 128;            // pixels per GEMM block
  const int TPB = HW / NPB;       // GEMM pixel tiles per image

  const float* x    = (const float*)d_in[0];
  const float* n1g  = (const float*)d_in[1];
  const float* n1b  = (const float*)d_in[2];
  const float* lw1  = (const float*)d_in[3];
  const float* lb1  = (const float*)d_in[4];
  const float* lw2  = (const float*)d_in[5];
  const float* lb2  = (const float*)d_in[6];
  const float* lwsc = (const float*)d_in[7];
  const float* lbsc = (const float*)d_in[8];
  const float* lw3  = (const float*)d_in[9];
  const float* lb3  = (const float*)d_in[10];
  const float* gw1  = (const float*)d_in[11];
  const float* gb1  = (const float*)d_in[12];
  const float* gw2  = (const float*)d_in[13];
  const float* gb2  = (const float*)d_in[14];
  const float* gwsc = (const float*)d_in[15];
  const float* gbsc = (const float*)d_in[16];
  const float* gw3  = (const float*)d_in[17];
  const float* gb3  = (const float*)d_in[18];
  const float* n2g  = (const float*)d_in[19];
  const float* n2b  = (const float*)d_in[20];
  const float* c1w  = (const float*)d_in[21];
  const float* c1b  = (const float*)d_in[22];
  const float* ciw1 = (const float*)d_in[23];
  const float* cib1 = (const float*)d_in[24];
  const float* cibg = (const float*)d_in[25];
  const float* cibb = (const float*)d_in[26];
  const float* ciw2 = (const float*)d_in[27];
  const float* cib2 = (const float*)d_in[28];
  const float* siw1 = (const float*)d_in[29];
  const float* sib1 = (const float*)d_in[30];
  const float* sibg = (const float*)d_in[31];
  const float* sibb = (const float*)d_in[32];
  const float* siw2 = (const float*)d_in[33];
  const float* sib2 = (const float*)d_in[34];
  const float* paw  = (const float*)d_in[35];
  const float* pab  = (const float*)d_in[36];
  const float* png  = (const float*)d_in[37];
  const float* pnb  = (const float*)d_in[38];
  const float* qkvw = (const float*)d_in[39];
  const float* qkvb = (const float*)d_in[40];
  const float* rpb  = (const float*)d_in[41];
  const float* n4g  = (const float*)d_in[42];
  const float* n4b  = (const float*)d_in[43];
  const float* c4w  = (const float*)d_in[44];
  const float* c4b  = (const float*)d_in[45];
  const float* n3g  = (const float*)d_in[46];
  const float* n3b  = (const float*)d_in[47];
  const float* c2w  = (const float*)d_in[48];
  const float* c2b  = (const float*)d_in[49];
  const float* c3w  = (const float*)d_in[50];
  const float* c3b  = (const float*)d_in[51];
  const float* beta = (const float*)d_in[52];
  const float* gamm = (const float*)d_in[53];

  char* ws = (char*)d_ws;
  size_t off = 0;
  auto take = [&](size_t bytes) { size_t r = off; off += (bytes + 255) & ~(size_t)255; return r; };
  bf16*  xnbf  = (bf16*) (ws + take(P * 64 * 2));
  bf16*  tbf   = (bf16*) (ws + take(P * 256 * 2));  // reused: qkv (192ch), c2out (128ch)
  bf16*  glbf  = (bf16*) (ws + take(P * 64 * 2));   // reused: paw out
  bf16*  ggbf  = (bf16*) (ws + take(P * 64 * 2));   // reused: xw (token-LN out)
  bf16*  cf0bf = (bf16*) (ws + take(P * 128 * 2));  // reused: ycat
  bf16*  cf0n  = (bf16*) (ws + take(P * 128 * 2));  // reused: ffn gate + yn
  float* cf32  = (float*)(ws + take(P * 64 * 4));
  float* xs32  = (float*)(ws + take(P * 64 * 4));
  float* y32   = (float*)(ws + take(P * 64 * 4));
  float* meanl = (float*)(ws + take(512 * 4));
  float* meang = (float*)(ws + take(512 * 4));
  float* scal  = (float*)(ws + take(512 * 4));
  float* scag  = (float*)(ws + take(512 * 4));
  float* cfmn  = (float*)(ws + take(512 * 4));
  float* cibuf = (float*)(ws + take(512 * 4));
  // pre-converted bf16 weights
  bf16* wb_lw1 = (bf16*)(ws + take(8192 * 2));
  bf16* wb_gw1 = (bf16*)(ws + take(8192 * 2));
  bf16* wb_lw3 = (bf16*)(ws + take(4096 * 2));
  bf16* wb_gw3 = (bf16*)(ws + take(4096 * 2));
  bf16* wb_c1  = (bf16*)(ws + take(8192 * 2));
  bf16* wb_paw = (bf16*)(ws + take(4096 * 2));
  bf16* wb_qkv = (bf16*)(ws + take(12288 * 2));
  bf16* wb_c4  = (bf16*)(ws + take(8192 * 2));
  bf16* wb_c2  = (bf16*)(ws + take(8192 * 2));
  bf16* wb_c3  = (bf16*)(ws + take(4096 * 2));

  bf16*  qkvbf = tbf;
  bf16*  pawbf = glbf;
  bf16*  xwbf  = ggbf;
  bf16*  ycat  = cf0bf;
  bf16*  c2out = tbf;
  bf16*  ffng  = cf0n;
  bf16*  ynbf  = cf0n + P * 64;

  auto shm = [NPB](int M, int K) { return (size_t)(M + NPB) * (K + 8) * 2; };
  const int LNG = (int)(P / 256);

  // 0) one-shot f32->bf16 weight conversion
  k_cvt<<<32, 256, 0, stream>>>(lw1, wb_lw1, 8192);
  k_cvt<<<32, 256, 0, stream>>>(gw1, wb_gw1, 8192);
  k_cvt<<<16, 256, 0, stream>>>(lw3, wb_lw3, 4096);
  k_cvt<<<16, 256, 0, stream>>>(gw3, wb_gw3, 4096);
  k_cvt<<<32, 256, 0, stream>>>(c1w, wb_c1, 8192);
  k_cvt<<<16, 256, 0, stream>>>(paw, wb_paw, 4096);
  k_cvt<<<48, 256, 0, stream>>>(qkvw, wb_qkv, 12288);
  k_cvt<<<32, 256, 0, stream>>>(c4w, wb_c4, 8192);
  k_cvt<<<32, 256, 0, stream>>>(c2w, wb_c2, 8192);
  k_cvt<<<16, 256, 0, stream>>>(c3w, wb_c3, 4096);

  // 1) xn = LN2d(x)
  k_lnch<float><<<LNG, 256, 0, stream>>>(x, nullptr, n1g, n1b, 1e-6f, 64, xnbf, HW);
  // 2) expand 1x1 for both branches -> t[0:128]=local, t[128:256]=global
  k_gemm<<<B * TPB, 256, shm(128, 64), stream>>>(wb_lw1, lb1, xnbf, nullptr, nullptr, tbf,
                                                 nullptr, nullptr, 128, 64, HW, TPB, 256, 0);
  k_gemm<<<B * TPB, 256, shm(128, 64), stream>>>(wb_gw1, gb1, xnbf, nullptr, nullptr, tbf,
                                                 nullptr, nullptr, 128, 64, HW, TPB, 256, 128);
  // 3) depthwise 3x3/5x5 + SimpleGate
  k_dwgate<<<dim3(HW / 256, 64, B), 256, 0, stream>>>(tbf, lw2, lb2, gw2, gb2, glbf, ggbf, H, W);
  // 4) spatial means for SCA
  k_rowmean<bf16><<<B * 64, 256, 0, stream>>>(glbf, meanl, HW);
  k_rowmean<bf16><<<B * 64, 256, 0, stream>>>(ggbf, meang, HW);
  // 5) SCA weights
  k_sca<<<B, 128, 0, stream>>>(lwsc, lbsc, gwsc, gbsc, meanl, meang, scal, scag);
  // 6) per-branch 1x1 with SCA scaling -> cf0 (concat, 128ch)
  k_gemm<<<B * TPB, 256, shm(64, 64), stream>>>(wb_lw3, lb3, glbf, scal, nullptr, cf0bf,
                                                nullptr, nullptr, 64, 64, HW, TPB, 128, 0);
  k_gemm<<<B * TPB, 256, shm(64, 64), stream>>>(wb_gw3, gb3, ggbf, scag, nullptr, cf0bf,
                                                nullptr, nullptr, 64, 64, HW, TPB, 128, 64);
  // 7) LN2d(n2) over 128ch
  k_lnch<bf16><<<LNG, 256, 0, stream>>>(cf0bf, nullptr, n2g, n2b, 1e-6f, 128, cf0n, HW);
  // 8) c1: 128 -> 64  -> cf (f32)
  k_gemm<<<B * TPB, 256, shm(64, 128), stream>>>(wb_c1, c1b, cf0n, nullptr, cf32, nullptr,
                                                 nullptr, nullptr, 64, 128, HW, TPB, 64, 0);
  // 9) channel interaction MLP on mean(cf)
  k_rowmean<float><<<B * 64, 256, 0, stream>>>(cf32, cfmn, HW);
  k_mlp_ci<<<B, 64, 0, stream>>>(cfmn, ciw1, cib1, cibg, cibb, ciw2, cib2, cibuf);
  // 10) attention pre-proj (paw) + token LN
  k_gemm<<<B * TPB, 256, shm(64, 64), stream>>>(wb_paw, pab, xnbf, nullptr, nullptr, pawbf,
                                                nullptr, nullptr, 64, 64, HW, TPB, 64, 0);
  k_lnch<bf16><<<LNG, 256, 0, stream>>>(pawbf, nullptr, png, pnb, 1e-5f, 64, xwbf, HW);
  // 11) qkv projection: 64 -> 192
  k_gemm<<<B * TPB, 256, shm(192, 64), stream>>>(wb_qkv, qkvb, xwbf, nullptr, nullptr, qkvbf,
                                                 nullptr, nullptr, 192, 64, HW, TPB, 192, 0);
  // 12) window MHA (8x8 windows, 4 heads, rel-pos bias, V*sigmoid(ci) folded into epilogue)
  k_attn<<<B * (H / 8) * (W / 8) * 2, 64, 55296, stream>>>(qkvbf, cibuf, rpb, xs32, H, W);
  // 13) spatial interaction: cf *= sigmoid(MLP(xs)) in place
  k_si<<<LNG, 256, 0, stream>>>(xs32, siw1, sib1, sibg, sibb, siw2, sib2, cf32, HW);
  // 14) LN2d(n4) over concat[cf, xs]
  k_lnch<float><<<LNG, 256, 0, stream>>>(cf32, xs32, n4g, n4b, 1e-6f, 128, ycat, HW);
  // 15) c4: 128 -> 64, y = x + beta * (...)
  k_gemm<<<B * TPB, 256, shm(64, 128), stream>>>(wb_c4, c4b, ycat, nullptr, y32, nullptr,
                                                 x, beta, 64, 128, HW, TPB, 64, 0);
  // 16) FFN: LN2d(n3) -> c2 (64->128) -> gate -> c3 (64->64), out = y + gamma * (...)
  k_lnch<float><<<LNG, 256, 0, stream>>>(y32, nullptr, n3g, n3b, 1e-6f, 64, ynbf, HW);
  k_gemm<<<B * TPB, 256, shm(128, 64), stream>>>(wb_c2, c2b, ynbf, nullptr, nullptr, c2out,
                                                 nullptr, nullptr, 128, 64, HW, TPB, 128, 0);
  k_gate<<<(int)(P * 64 / 256), 256, 0, stream>>>(c2out, ffng, HW);
  k_gemm<<<B * TPB, 256, shm(64, 64), stream>>>(wb_c3, c3b, ffng, nullptr, (float*)d_out, nullptr,
                                                y32, gamm, 64, 64, HW, TPB, 64, 0);
}
